// GAT_17016660427563
// MI455X (gfx1250) — matrix operand order repaired and checked by reading, not compiled, not run on hardware
//
#include <hip/hip_runtime.h>
#include <hip/hip_bf16.h>
#include <math.h>

// ---------------------------------------------------------------------------
// GAT layer for MI455X (gfx1250, wave32, WMMA + async global->LDS)
//   N=50000 nodes, E=800000 edges, F_IN=256, H=4 heads, D=16  (H*D = 64)
//
// Pipeline:
//   k_convert : feat fp32 -> bf16            (featB [N,256])
//   k_packw   : W = [W_src;W_dst] -> bf16    (Wb [128,256]) + bias concat
//   k_init    : m=-inf, s=0, out=0
//   k_gemm    : featP[N,128] = featB @ Wb^T + bias
//               (GLOBAL_LOAD_ASYNC_TO_LDS_B128 A staging with bank-conflict-
//                free padded LDS layout; 80-row M tile -> 40 WMMA per wave)
//   k_score   : score[e,h] = attn[h]·leakyrelu(el+er); atomicMax -> m[dst,h]
//   k_exp     : a = exp(score - m[dst]); atomicAdd -> s[dst,h]
//   k_agg     : out[dst,h,:] += el * (a / s[dst,h])  (atomic scatter-add)
// ---------------------------------------------------------------------------

#define NN     50000
#define EE     800000
#define FIN    256
#define HH     4
#define DD     16
#define HD     64      // H*D
#define COLS   128     // 2*H*D  (src proj | dst proj)
#define MT     80      // GEMM rows per block (50000 = 625 * 80, exact)
#define LDAP   264     // padded LDS row stride in bf16 elements:
                       // 132 dwords == 4 (mod 64 banks) -> lane r hits banks
                       // 4r..4r+3: conflict-free 16-lane fragment reads
#define NEG_SLOPE 0.2f

typedef __attribute__((ext_vector_type(16))) __bf16 v16bf;
typedef __attribute__((ext_vector_type(8)))  float  v8f;
typedef __attribute__((ext_vector_type(4)))  unsigned int v4u;

// ---------------------------------------------------------------- convert ---
__global__ __launch_bounds__(256)
void k_convert(const float* __restrict__ feat, __bf16* __restrict__ featB, int n) {
    int t = blockIdx.x * blockDim.x + threadIdx.x;
    if (t < n) featB[t] = (__bf16)feat[t];
}

// ------------------------------------------------------------------ packw ---
__global__ __launch_bounds__(256)
void k_packw(const float* __restrict__ Wsrc, const float* __restrict__ Wdst,
             const float* __restrict__ bsrc, const float* __restrict__ bdst,
             __bf16* __restrict__ Wb, float* __restrict__ bias) {
    int t = blockIdx.x * blockDim.x + threadIdx.x;  // 0 .. 32767
    if (t < COLS * FIN) {
        int r = t / FIN, c = t % FIN;
        float w = (r < HD) ? Wsrc[r * FIN + c] : Wdst[(r - HD) * FIN + c];
        Wb[t] = (__bf16)w;
    }
    if (t < COLS) bias[t] = (t < HD) ? bsrc[t] : bdst[t - HD];
}

// ------------------------------------------------------------------- init ---
__global__ __launch_bounds__(256)
void k_init(float* __restrict__ m, float* __restrict__ s, float* __restrict__ out) {
    int t = blockIdx.x * blockDim.x + threadIdx.x;
    if (t < NN * HD) out[t] = 0.0f;                 // 3.2M
    if (t < NN * HH) { m[t] = -INFINITY; s[t] = 0.0f; }
}

// ------------------------------------------------------------------- gemm ---
// featP[N,128] = featB[N,256] @ Wb[128,256]^T + bias
// One block = 80 output rows (5 x 16); 8 waves, each owns one 16-wide column
// tile and reuses its B fragment across the 5 M-subtiles (40 WMMA/wave).
// A-tile staged in padded LDS via GLOBAL_LOAD_ASYNC_TO_LDS_B128 (ASYNCcnt
// path: no VGPR round-trip).  lA's address is taken via ptrtoint so the LDS
// object escapes into the asm -> compiler must honor the asm's LDS writes.
__global__ __launch_bounds__(256)
void k_gemm(const __bf16* __restrict__ A,    // [N,256]
            const __bf16* __restrict__ W,    // [128,256] row-major
            const float*  __restrict__ bias, // [128]
            float* __restrict__ out)         // [N,128]
{
    __shared__ __bf16 lA[MT * LDAP];         // 42240 bytes, padded rows
    const int lane  = threadIdx.x & 31;
    const int wave  = threadIdx.x >> 5;      // 0..7 -> column tile
    const int mtile = blockIdx.x;            // 0..624

    // ---- async stage: A tile (80 x 256 bf16 = 2560 x b128 chunks) -> LDS.
    //      chunk c covers global row c>>5, 16B-column c&31; LDS dest is the
    //      padded row.  256 threads x 10 chunks.
    {
        const unsigned lbase = (unsigned)(unsigned long long)&lA[0]; // LDS off (escapes lA)
        const unsigned long long gbase =
            (unsigned long long)(const char*)(A + (size_t)mtile * MT * FIN);
        #pragma unroll
        for (int i = 0; i < 10; ++i) {
            unsigned c    = (unsigned)threadIdx.x + i * 256u;        // chunk id
            unsigned loff = lbase + (c >> 5) * (LDAP * 2u) + (c & 31u) * 16u;
            unsigned long long gaddr = gbase + (unsigned long long)c * 16u;
            asm volatile("global_load_async_to_lds_b128 %0, %1, off"
                         :: "v"(loff), "v"(gaddr) : "memory");
        }
        asm volatile("s_wait_asynccnt 0x0" ::: "memory");
    }
    __syncthreads();

    const int row   = lane & 15;             // A row within subtile / B column
    const int khalf = lane >> 4;             // 0 or 1 (lane-group K split)
    const __bf16* wrow = W + (size_t)(wave * 16 + row) * FIN;

    v8f acc[5] = {};
    #pragma unroll
    for (int k0 = 0; k0 < FIN; k0 += 32) {
        // B fragment (32x16; column-major B == row-major W):
        //   lanes 0-15 : K = k0+{0..15};  lanes 16-31: K = k0+{16..31}
        v16bf b;
        const __bf16* bp = wrow + k0 + khalf * 16;
        ((v4u*)&b)[0] = ((const v4u*)bp)[0];
        ((v4u*)&b)[1] = ((const v4u*)bp)[1];

        #pragma unroll
        for (int ms = 0; ms < 5; ++ms) {
            // A fragment (K order matches B):
            //   lanes 0-15 : K = k0+{0..15}
            //   lanes 16-31: K = k0+{16..31}
            v16bf a;
            const __bf16* ap = lA + (ms * 16 + row) * LDAP + k0 + khalf * 16;
            ((v4u*)&a)[0] = *(const v4u*)ap;
            ((v4u*)&a)[1] = *(const v4u*)(ap + 8);

            acc[ms] = __builtin_amdgcn_wmma_f32_16x16x32_bf16(
                          false, a, false, b, (short)0, acc[ms], false, false);
        }
    }

    // C/D layout: lane L, vgpr v -> m = v + 8*(L>>4), n = L&15
    const int n = wave * 16 + row;
    const float bn = bias[n];
    #pragma unroll
    for (int ms = 0; ms < 5; ++ms) {
        #pragma unroll
        for (int v = 0; v < 8; ++v) {
            int mm = ms * 16 + v + 8 * khalf;
            out[((size_t)mtile * MT + mm) * COLS + n] = acc[ms][v] + bn;
        }
    }
}

// ------------------------------------------------------------------ score ---
// One thread per (edge, head).  el = featP[src][h*16..], er = featP[dst][64+h*16..]
__global__ __launch_bounds__(256)
void k_score(const int* __restrict__ src, const int* __restrict__ dst,
             const float* __restrict__ featP, const float* __restrict__ attn,
             float* __restrict__ score, float* __restrict__ m)
{
    int t = blockIdx.x * blockDim.x + threadIdx.x;
    if (t >= EE * HH) return;
    int e = t >> 2, h = t & 3;
    int si = src[e], di = dst[e];
    const float4* el = (const float4*)(featP + (size_t)si * COLS + h * DD);
    const float4* er = (const float4*)(featP + (size_t)di * COLS + HD + h * DD);
    const float4* at = (const float4*)(attn + h * DD);
    float sc = 0.0f;
    #pragma unroll
    for (int j = 0; j < 4; ++j) {
        float4 a = el[j], b = er[j], w = at[j];
        float v0 = a.x + b.x, v1 = a.y + b.y, v2 = a.z + b.z, v3 = a.w + b.w;
        v0 = v0 > 0.0f ? v0 : NEG_SLOPE * v0;
        v1 = v1 > 0.0f ? v1 : NEG_SLOPE * v1;
        v2 = v2 > 0.0f ? v2 : NEG_SLOPE * v2;
        v3 = v3 > 0.0f ? v3 : NEG_SLOPE * v3;
        sc += v0 * w.x + v1 * w.y + v2 * w.z + v3 * w.w;
    }
    score[t] = sc;
    atomicMax(&m[di * HH + h], sc);   // -> global float max atomic
}

// -------------------------------------------------------------------- exp ---
__global__ __launch_bounds__(256)
void k_exp(const int* __restrict__ dst, float* __restrict__ score,
           const float* __restrict__ m, float* __restrict__ s)
{
    int t = blockIdx.x * blockDim.x + threadIdx.x;
    if (t >= EE * HH) return;
    int e = t >> 2, h = t & 3;
    int di = dst[e];
    float a = __expf(score[t] - m[di * HH + h]);
    score[t] = a;
    atomicAdd(&s[di * HH + h], a);
}

// -------------------------------------------------------------------- agg ---
// One thread per (edge, head, d-quad): 16 threads/edge, float4 gather,
// 4x atomicAdd scatter into out[dst].
__global__ __launch_bounds__(256)
void k_agg(const int* __restrict__ src, const int* __restrict__ dst,
           const float* __restrict__ featP, const float* __restrict__ score,
           const float* __restrict__ s, float* __restrict__ out)
{
    int t = blockIdx.x * blockDim.x + threadIdx.x;
    if (t >= EE * 16) return;
    int e = t >> 4, q = t & 15;
    int h = q >> 2, dq = q & 3;
    int si = src[e], di = dst[e];
    float alpha = score[e * HH + h] / s[di * HH + h];
    float4 el = *(const float4*)(featP + (size_t)si * COLS + h * DD + dq * 4);
    float* o = out + (size_t)di * HD + h * DD + dq * 4;
    atomicAdd(o + 0, el.x * alpha);
    atomicAdd(o + 1, el.y * alpha);
    atomicAdd(o + 2, el.z * alpha);
    atomicAdd(o + 3, el.w * alpha);
}

// ---------------------------------------------------------------- launch ---
static inline size_t alignup(size_t x) { return (x + 255) & ~(size_t)255; }

extern "C" void kernel_launch(void* const* d_in, const int* in_sizes, int n_in,
                              void* d_out, int out_size, void* d_ws, size_t ws_size,
                              hipStream_t stream) {
    const float* feat  = (const float*)d_in[0];
    const int*   src   = (const int*)  d_in[1];
    const int*   dst   = (const int*)  d_in[2];
    const float* Wsrc  = (const float*)d_in[3];
    const float* bsrc  = (const float*)d_in[4];
    const float* Wdst  = (const float*)d_in[5];
    const float* bdst  = (const float*)d_in[6];
    const float* attn  = (const float*)d_in[7];
    float* out = (float*)d_out;

    // workspace carve-up (~66 MB)
    char* ws = (char*)d_ws;
    size_t o = 0;
    __bf16* featB = (__bf16*)(ws + o); o = alignup(o + (size_t)NN * FIN * 2);
    __bf16* Wb    = (__bf16*)(ws + o); o = alignup(o + (size_t)COLS * FIN * 2);
    float*  bias  = (float*)(ws + o);  o = alignup(o + COLS * 4);
    float*  featP = (float*)(ws + o);  o = alignup(o + (size_t)NN * COLS * 4);
    float*  score = (float*)(ws + o);  o = alignup(o + (size_t)EE * HH * 4);
    float*  mbuf  = (float*)(ws + o);  o = alignup(o + (size_t)NN * HH * 4);
    float*  sbuf  = (float*)(ws + o);  o = alignup(o + (size_t)NN * HH * 4);
    (void)ws_size; (void)in_sizes; (void)n_in; (void)out_size;

    const int B = 256;
    k_convert<<<(NN * FIN + B - 1) / B, B, 0, stream>>>(feat, featB, NN * FIN);
    k_packw  <<<(COLS * FIN + B - 1) / B, B, 0, stream>>>(Wsrc, Wdst, bsrc, bdst, Wb, bias);
    k_init   <<<(NN * HD + B - 1) / B, B, 0, stream>>>(mbuf, sbuf, out);
    k_gemm   <<<NN / MT, B, 0, stream>>>(featB, Wb, bias, featP);
    k_score  <<<(EE * HH + B - 1) / B, B, 0, stream>>>(src, dst, featP, attn, score, mbuf);
    k_exp    <<<(EE * HH + B - 1) / B, B, 0, stream>>>(dst, score, mbuf, sbuf);
    k_agg    <<<(EE * 16 + B - 1) / B, B, 0, stream>>>(src, dst, featP, score, sbuf, out);
}
